// MultiheadwRotataryEmbedding_9380208574756
// MI455X (gfx1250) — matrix operand order repaired and checked by reading, not compiled, not run on hardware
//
#include <hip/hip_runtime.h>

#define B_DIM  4
#define T_DIM  2048
#define D_DIM  1024
#define H_DIM  16
#define DH_DIM 64

typedef __bf16 v16bf __attribute__((ext_vector_type(16)));
typedef float  v8f   __attribute__((ext_vector_type(8)));
typedef int    v4i   __attribute__((ext_vector_type(4)));

union FragU { unsigned u[8]; v16bf v; };

// ---- CDNA5 async global->LDS path (guarded: falls back to sync staging) ----
#if defined(__AMDGCN__) && defined(__has_builtin)
#if __has_builtin(__builtin_amdgcn_global_load_async_to_lds_b128) && \
    __has_builtin(__builtin_amdgcn_s_wait_asynccnt)
#define USE_ASYNC_LDS 1
#endif
#endif

__device__ __forceinline__ unsigned pack_bf16(float x, float y) {
  union { __bf16 h[2]; unsigned u; } p;
  p.h[0] = (__bf16)x; p.h[1] = (__bf16)y;
  return p.u;
}

// 16-bit A-matrix 16x32 layout (ISA 7.12.2): VGPR v, lane-half kh -> packed-pair k offset
__device__ __forceinline__ int a_koff(int v, int kh) {
  return ((v >> 2) << 4) | (kh << 3) | ((v & 3) << 1);
}
// 16-bit B-matrix 32x16 layout: same k order as A (VGPR v, lane-half kh -> packed-pair k offset)
__device__ __forceinline__ int b_koff(int v, int kh) {
  return ((v >> 2) << 4) | (kh << 3) | ((v & 3) << 1);
}

__device__ __forceinline__ void load_afrag(FragU& f, const __bf16* __restrict__ arow,
                                           int kk, int kh) {
#pragma unroll
  for (int v = 0; v < 8; ++v)
    f.u[v] = *(const unsigned*)(arow + kk + a_koff(v, kh));
}
__device__ __forceinline__ void load_bfrag(FragU& f, const __bf16* __restrict__ wr,
                                           int kk, int kh) {
#pragma unroll
  for (int v = 0; v < 8; ++v)
    f.u[v] = *(const unsigned*)(wr + kk + b_koff(v, kh));
}
__device__ __forceinline__ void mma4(v8f* acc, const FragU& a, const FragU* b) {
#pragma unroll
  for (int nt = 0; nt < 4; ++nt)
    acc[nt] = __builtin_amdgcn_wmma_f32_16x16x32_bf16(
        false, a.v, false, b[nt].v, (short)0, acc[nt], false, false);
}

// Software-pipelined mainloop: fragments for step k+1 are loaded before the
// WMMA chain of step k, so VMEM latency overlaps matrix math. Manually
// unrolled x2 so all fragment register indices are static. K % 64 == 0.
__device__ __forceinline__ void gemm_mainloop(const __bf16* __restrict__ arow,
                                              const __bf16* const* __restrict__ wrow,
                                              int K, int kh, v8f* acc) {
  FragU a0, a1, b0[4], b1[4];
  load_afrag(a0, arow, 0, kh);
#pragma unroll
  for (int nt = 0; nt < 4; ++nt) load_bfrag(b0[nt], wrow[nt], 0, kh);
  for (int kk = 0; kk < K; kk += 64) {
    load_afrag(a1, arow, kk + 32, kh);
#pragma unroll
    for (int nt = 0; nt < 4; ++nt) load_bfrag(b1[nt], wrow[nt], kk + 32, kh);
    mma4(acc, a0, b0);
    if (kk + 64 < K) {
      load_afrag(a0, arow, kk + 64, kh);
#pragma unroll
      for (int nt = 0; nt < 4; ++nt) load_bfrag(b0[nt], wrow[nt], kk + 64, kh);
    }
    mma4(acc, a1, b1);
  }
}

// ---------------------------------------------------------------------------
// f32 -> bf16 bulk conversion (one pass; makes all GEMM loops pure bf16 loads)
// ---------------------------------------------------------------------------
__global__ __launch_bounds__(256) void cvt_bf16(const float* __restrict__ in,
                                                __bf16* __restrict__ out, int n4) {
  int i = blockIdx.x * 256 + threadIdx.x;
  if (i < n4) {
    float4 f = ((const float4*)in)[i];
    uint2 r;
    r.x = pack_bf16(f.x, f.y);
    r.y = pack_bf16(f.z, f.w);
    ((uint2*)out)[i] = r;
  }
}

// ---------------------------------------------------------------------------
// C[M,N] = A[M,K] @ W[N,K]^T + bias, all-bf16 operands, bf16 out,
// optional fused RoPE epilogue (64-wide N tile == exactly one head).
// One wave: 16(M) x 64(N) tile, K-loop step 32, 4 WMMA / step.
// ---------------------------------------------------------------------------
template <bool DO_ROPE>
__global__ __launch_bounds__(128) void gemm_rope(
    const __bf16* __restrict__ A, const __bf16* __restrict__ W,
    const float* __restrict__ bias, __bf16* __restrict__ out,
    int M, int N, int K)
{
  const int wave = threadIdx.x >> 5;
  const int lane = threadIdx.x & 31;
  const int ln = lane & 15, kh = lane >> 4;
  const int m0 = blockIdx.x * 64 + wave * 16;
  const int n0 = blockIdx.y * 64;

  const __bf16* arow = A + (size_t)(m0 + ln) * K;
  const __bf16* wrow[4];
#pragma unroll
  for (int nt = 0; nt < 4; ++nt)
    wrow[nt] = W + (size_t)(n0 + nt * 16 + ln) * K;

  v8f acc[4] = {};
  gemm_mainloop(arow, wrow, K, kh, acc);

  if constexpr (DO_ROPE) {
    __shared__ float lds_t[4][16 * 64];
    float* t = lds_t[wave];
#pragma unroll
    for (int nt = 0; nt < 4; ++nt) {
      float bv = bias[n0 + nt * 16 + ln];
#pragma unroll
      for (int v = 0; v < 8; ++v)
        t[(kh * 8 + v) * 64 + nt * 16 + ln] = acc[nt][v] + bv;
    }
    __syncthreads();
    for (int p = lane; p < 512; p += 32) {
      int r = p >> 5;                 // tile row 0..15
      int d = p & 31;                 // rotary pair 0..31
      int trow = (m0 + r) % T_DIM;    // token position
      float ang = (float)trow * __powf(10000.0f, -(float)d * (1.0f / 32.0f));
      float sn, cs;
      __sincosf(ang, &sn, &cs);
      float x1 = t[r * 64 + 2 * d];
      float x2 = t[r * 64 + 2 * d + 1];
      __bf16* orow = out + (size_t)(m0 + r) * N + n0;
      orow[d]      = (__bf16)(x1 * cs - x2 * sn);
      orow[d + 32] = (__bf16)(x1 * sn + x2 * cs);
    }
  } else {
#pragma unroll
    for (int nt = 0; nt < 4; ++nt) {
      int n = n0 + nt * 16 + ln;
      float bv = bias[n];
#pragma unroll
      for (int v = 0; v < 8; ++v)
        out[(size_t)(m0 + kh * 8 + v) * N + n] = (__bf16)(acc[nt][v] + bv);
    }
  }
}

// ---------------------------------------------------------------------------
// Final projection: bf16 A and W, f32 out + bias.
// ---------------------------------------------------------------------------
__global__ __launch_bounds__(128) void gemm_final(
    const __bf16* __restrict__ A, const __bf16* __restrict__ W,
    const float* __restrict__ bias, float* __restrict__ out,
    int M, int N, int K)
{
  const int wave = threadIdx.x >> 5;
  const int lane = threadIdx.x & 31;
  const int ln = lane & 15, kh = lane >> 4;
  const int m0 = blockIdx.x * 64 + wave * 16;
  const int n0 = blockIdx.y * 64;

  const __bf16* arow = A + (size_t)(m0 + ln) * K;
  const __bf16* wrow[4];
#pragma unroll
  for (int nt = 0; nt < 4; ++nt)
    wrow[nt] = W + (size_t)(n0 + nt * 16 + ln) * K;

  v8f acc[4] = {};
  gemm_mainloop(arow, wrow, K, kh, acc);

#pragma unroll
  for (int nt = 0; nt < 4; ++nt) {
    int n = n0 + nt * 16 + ln;
    float bv = bias[n];
#pragma unroll
    for (int v = 0; v < 8; ++v)
      out[(size_t)(m0 + kh * 8 + v) * N + n] = acc[nt][v] + bv;
  }
}

// ---------------------------------------------------------------------------
// Cooperative flash attention: one block (4 waves) = one (b,h) x 64 queries.
// K/V 32-key tiles staged in LDS once per block (async, double-buffered),
// shared by all 4 waves. 8 WMMA per wave per 32-key tile.
// ---------------------------------------------------------------------------
__device__ __forceinline__ void stage_kv(const __bf16* __restrict__ Kbase,
                                         const __bf16* __restrict__ Vbase,
                                         __bf16* Kt, __bf16* Vt,
                                         int s0, int tid) {
  // 32 rows x 64 bf16 (128B) per matrix -> 256 x 16B segments each
  for (int i = tid; i < 256; i += 128) {
    int row = i >> 3, seg = (i & 7) * 8;
    const __bf16* ks = Kbase + (size_t)(s0 + row) * D_DIM + seg;
    const __bf16* vs = Vbase + (size_t)(s0 + row) * D_DIM + seg;
    __bf16* kd = Kt + row * 64 + seg;
    __bf16* vd = Vt + row * 64 + seg;
#ifdef USE_ASYNC_LDS
    // per-lane b128 async copy, ASYNCcnt-tracked (CDNA5 ISA)
    __builtin_amdgcn_global_load_async_to_lds_b128((v4i*)ks, (v4i*)kd, 0, 0);
    __builtin_amdgcn_global_load_async_to_lds_b128((v4i*)vs, (v4i*)vd, 0, 0);
#else
    *(float4*)kd = *(const float4*)ks;
    *(float4*)vd = *(const float4*)vs;
#endif
  }
}

__global__ __launch_bounds__(128) void flash_attn(
    const __bf16* __restrict__ Q, const __bf16* __restrict__ Km,
    const __bf16* __restrict__ V, __bf16* __restrict__ O)
{
  __shared__ __align__(16) __bf16 Kt[2][32 * 64];   // double-buffered key tile
  __shared__ __align__(16) __bf16 Vt[2][32 * 64];   // double-buffered value tile
  __shared__ __bf16 Pt[4][16 * 32];                 // per-wave P transpose staging

  const int tid  = threadIdx.x;
  const int wave = tid >> 5;
  const int lane = tid & 31;
  const int ln = lane & 15, kh = lane >> 4;

  const int qblk = blockIdx.x & 31;          // T/64 = 32 query blocks
  const int h    = (blockIdx.x >> 5) & 15;
  const int b    = blockIdx.x >> 9;
  const int q0b  = qblk * 64;
  const int q0   = q0b + wave * 16;
  const size_t baseBT = (size_t)b * T_DIM;
  const __bf16* Kbase = Km + baseBT * D_DIM + h * DH_DIM;
  const __bf16* Vbase = V  + baseBT * D_DIM + h * DH_DIM;

  // Q fragments (16 x 64, A-layout, two K=32 halves), resident all pass
  const __bf16* qrow = Q + (baseBT + q0 + ln) * D_DIM + h * DH_DIM;
  FragU aq[2];
#pragma unroll
  for (int j = 0; j < 2; ++j)
#pragma unroll
    for (int v = 0; v < 8; ++v)
      aq[j].u[v] = *(const unsigned*)(qrow + j * 32 + a_koff(v, kh));

  v8f o_acc[4] = {};
  float m_i[8], l_i[8];
#pragma unroll
  for (int v = 0; v < 8; ++v) { m_i[v] = -3.0e38f; l_i[v] = 0.0f; }

  const int qlast = q0 + 15;
  const int s_end = q0b + 64;
  __bf16* pt = Pt[wave];

  // prologue: fill buffer 0
  stage_kv(Kbase, Vbase, Kt[0], Vt[0], 0, tid);
#ifdef USE_ASYNC_LDS
  __builtin_amdgcn_s_wait_asynccnt(0);
#endif
  __syncthreads();

  int cur = 0;
  for (int s0 = 0; s0 < s_end; s0 += 32, cur ^= 1) {
    // kick off async fill of the next tile while computing this one
    if (s0 + 32 < s_end)
      stage_kv(Kbase, Vbase, Kt[cur ^ 1], Vt[cur ^ 1], s0 + 32, tid);

    if (s0 <= qlast) {   // wave-uniform: waves fully past this key tile skip
      const __bf16* Kc = Kt[cur];
      const __bf16* Vc = Vt[cur];
      // S = Q @ K^T for 32 keys -> two 16x16 C tiles (frags from LDS)
      v8f s_acc[2] = {};
#pragma unroll
      for (int t2 = 0; t2 < 2; ++t2) {
        const __bf16* krow = Kc + (t2 * 16 + ln) * 64;
#pragma unroll
        for (int j = 0; j < 2; ++j) {
          FragU bk;
#pragma unroll
          for (int v = 0; v < 8; ++v)
            bk.u[v] = *(const unsigned*)(krow + j * 32 + b_koff(v, kh));
          s_acc[t2] = __builtin_amdgcn_wmma_f32_16x16x32_bf16(
              false, aq[j].v, false, bk.v, (short)0, s_acc[t2], false, false);
        }
      }
      // scale + causal mask + online softmax (rows live per 16-lane half)
      float corr[8], p0[8], p1[8];
#pragma unroll
      for (int v = 0; v < 8; ++v) {
        int mrow = q0 + kh * 8 + v;
        float s0v = (s0 + ln      <= mrow) ? s_acc[0][v] * 0.125f : -3.0e38f;
        float s1v = (s0 + 16 + ln <= mrow) ? s_acc[1][v] * 0.125f : -3.0e38f;
        float mx = fmaxf(s0v, s1v);
#pragma unroll
        for (int off = 1; off < 16; off <<= 1)
          mx = fmaxf(mx, __shfl_xor(mx, off, 32));
        float mn = fmaxf(m_i[v], mx);
        float cr = __expf(m_i[v] - mn);
        float e0 = __expf(s0v - mn);
        float e1 = __expf(s1v - mn);
        float rs = e0 + e1;
#pragma unroll
        for (int off = 1; off < 16; off <<= 1)
          rs += __shfl_xor(rs, off, 32);
        l_i[v] = l_i[v] * cr + rs;
        m_i[v] = mn;
        corr[v] = cr;
        p0[v] = e0;
        p1[v] = e1;
      }
      // rescale O; transpose P (C-layout -> A-layout) through per-wave LDS tile
#pragma unroll
      for (int v = 0; v < 8; ++v) {
        int r = kh * 8 + v;
        pt[r * 32 + ln]      = (__bf16)p0[v];
        pt[r * 32 + 16 + ln] = (__bf16)p1[v];
#pragma unroll
        for (int nt = 0; nt < 4; ++nt)
          o_acc[nt][v] *= corr[v];
      }
      v16bf apv;
#pragma unroll
      for (int v = 0; v < 8; ++v) {
        int k = a_koff(v, kh);
        apv[2 * v]     = pt[ln * 32 + k];
        apv[2 * v + 1] = pt[ln * 32 + k + 1];
      }
      // O += P @ V (V B-layout: packed pair spans two adjacent key rows)
#pragma unroll
      for (int nt = 0; nt < 4; ++nt) {
        v16bf bvv;
#pragma unroll
        for (int v = 0; v < 8; ++v) {
          int k = b_koff(v, kh);
          bvv[2 * v]     = Vc[k * 64 + nt * 16 + ln];
          bvv[2 * v + 1] = Vc[(k + 1) * 64 + nt * 16 + ln];
        }
        o_acc[nt] = __builtin_amdgcn_wmma_f32_16x16x32_bf16(
            false, apv, false, bvv, (short)0, o_acc[nt], false, false);
      }
    }

#ifdef USE_ASYNC_LDS
    __builtin_amdgcn_s_wait_asynccnt(0);   // next buffer resident
#endif
    __syncthreads();                        // all waves done with current buffer
  }

  // normalize and store bf16
#pragma unroll
  for (int v = 0; v < 8; ++v) {
    float inv = 1.0f / l_i[v];
    size_t row = (baseBT + q0 + kh * 8 + v) * D_DIM + h * DH_DIM;
#pragma unroll
    for (int nt = 0; nt < 4; ++nt)
      O[row + nt * 16 + ln] = (__bf16)(o_acc[nt][v] * inv);
  }
}

extern "C" void kernel_launch(void* const* d_in, const int* in_sizes, int n_in,
                              void* d_out, int out_size, void* d_ws, size_t ws_size,
                              hipStream_t stream) {
  (void)in_sizes; (void)n_in; (void)out_size; (void)ws_size;
  const float* x   = (const float*)d_in[0];
  const float* W_q = (const float*)d_in[1];
  const float* b_q = (const float*)d_in[2];
  const float* W_k = (const float*)d_in[3];
  const float* b_k = (const float*)d_in[4];
  const float* W_v = (const float*)d_in[5];
  const float* b_v = (const float*)d_in[6];
  const float* W_o = (const float*)d_in[7];
  const float* b_o = (const float*)d_in[8];
  float* out = (float*)d_out;

  const int M = B_DIM * T_DIM;            // 8192
  const int N = D_DIM, K = D_DIM;
  const size_t szMD = (size_t)M * D_DIM;  // 8M elements
  const size_t szW  = (size_t)D_DIM * D_DIM;

  __bf16* xb  = (__bf16*)d_ws;            // bf16 staging of inputs
  __bf16* Wqb = xb  + szMD;
  __bf16* Wkb = Wqb + szW;
  __bf16* Wvb = Wkb + szW;
  __bf16* Wob = Wvb + szW;
  __bf16* Qb  = Wob + szW;                // bf16 intermediates
  __bf16* Kb  = Qb  + szMD;
  __bf16* Vb  = Kb  + szMD;
  __bf16* Ob  = Vb  + szMD;

  // one-pass f32 -> bf16 conversions
  cvt_bf16<<<(int)(szMD / 4 + 255) / 256, 256, 0, stream>>>(x,   xb,  (int)(szMD / 4));
  cvt_bf16<<<(int)(szW  / 4 + 255) / 256, 256, 0, stream>>>(W_q, Wqb, (int)(szW / 4));
  cvt_bf16<<<(int)(szW  / 4 + 255) / 256, 256, 0, stream>>>(W_k, Wkb, (int)(szW / 4));
  cvt_bf16<<<(int)(szW  / 4 + 255) / 256, 256, 0, stream>>>(W_v, Wvb, (int)(szW / 4));
  cvt_bf16<<<(int)(szW  / 4 + 255) / 256, 256, 0, stream>>>(W_o, Wob, (int)(szW / 4));

  dim3 gg(M / 64, N / 64), bb(128);
  gemm_rope<true><<<gg, bb, 0, stream>>>(xb, Wqb, b_q, Qb, M, N, K);
  gemm_rope<true><<<gg, bb, 0, stream>>>(xb, Wkb, b_k, Kb, M, N, K);
  gemm_rope<false><<<gg, bb, 0, stream>>>(xb, Wvb, b_v, Vb, M, N, K);
  flash_attn<<<dim3(B_DIM * H_DIM * (T_DIM / 64)), bb, 0, stream>>>(Qb, Kb, Vb, Ob);
  gemm_final<<<gg, bb, 0, stream>>>(Ob, Wob, b_o, out, M, N, K);
}
